// MultiBasisBlock_27676769255894
// MI455X (gfx1250) — compile-verified, matrix operand order checked by reading, not run on hardware
//
#include <hip/hip_runtime.h>
#include <hip/hip_bf16.h>
#include <cstdint>

// ---------------------------------------------------------------------------
// CDNA5 (gfx1250) wave32 WMMA types
// ---------------------------------------------------------------------------
typedef __attribute__((ext_vector_type(16))) __bf16 v16bf;
typedef __attribute__((ext_vector_type(8)))  float  v8f;
typedef __attribute__((ext_vector_type(4)))  int    v4i;

#define WMMA_BF16(a, b, c) \
  __builtin_amdgcn_wmma_f32_16x16x32_bf16(false, (a), false, (b), (short)0, (c), false, false)

// Load a 16x32 bf16 A/B fragment for this lane from a row/col base pointer.
// Per ISA 7.12.2: elems[0..7] = K kb..kb+7, elems[8..15] = K kb+16..kb+23,
// kb = (lane>=16) ? 8 : 0.  Caller passes base+kb (16B aligned).
__device__ __forceinline__ v16bf ld_frag(const __bf16* p) {
  union { v4i q[2]; v16bf f; } u;
  u.q[0] = *(const v4i*)(p);
  u.q[1] = *(const v4i*)(p + 16);
  return u.f;
}

// Async global->LDS copy, 16B per lane, tracked by ASYNCcnt (ISA 15.18.3 #98).
// The DS-relative LDS address is the low 32 bits of the flat shared address
// (ISA 10.2).
__device__ __forceinline__ void async_ld16(const __bf16* g, __bf16* l) {
  const uint32_t lds = (uint32_t)(uintptr_t)l;
  const uint64_t ga  = (uint64_t)(uintptr_t)g;
  asm volatile("global_load_async_to_lds_b128 %0, %1, off"
               :: "v"(lds), "v"(ga) : "memory");
}
#define WAIT_ASYNC(n) asm volatile("s_wait_asynccnt " #n ::: "memory")

__device__ __forceinline__ float rmax16(float v) {
#pragma unroll
  for (int m = 1; m < 16; m <<= 1) v = fmaxf(v, __shfl_xor(v, m, 32));
  return v;
}
__device__ __forceinline__ float radd16(float v) {
#pragma unroll
  for (int m = 1; m < 16; m <<= 1) v += __shfl_xor(v, m, 32);
  return v;
}

// ---------------------------------------------------------------------------
// 1) Basis combination: W = sum_k alpha[layer,k] * base_k  -> bf16 [N,K] rows
// ---------------------------------------------------------------------------
__global__ __launch_bounds__(256) void combine4(
    const float* __restrict__ bases, const float* __restrict__ alpha,
    const int* __restrict__ layer, __bf16* __restrict__ out, int n) {
  const size_t i = ((size_t)blockIdx.x * 256 + threadIdx.x) * 4;
  if (i >= (size_t)n) return;
  const int L = *layer;
  const float a0 = alpha[L * 4 + 0], a1 = alpha[L * 4 + 1];
  const float a2 = alpha[L * 4 + 2], a3 = alpha[L * 4 + 3];
  const float4 b0 = *(const float4*)(bases + i);
  const float4 b1 = *(const float4*)(bases + (size_t)n + i);
  const float4 b2 = *(const float4*)(bases + 2 * (size_t)n + i);
  const float4 b3 = *(const float4*)(bases + 3 * (size_t)n + i);
  out[i + 0] = (__bf16)(a0 * b0.x + a1 * b1.x + a2 * b2.x + a3 * b3.x);
  out[i + 1] = (__bf16)(a0 * b0.y + a1 * b1.y + a2 * b2.y + a3 * b3.y);
  out[i + 2] = (__bf16)(a0 * b0.z + a1 * b1.z + a2 * b2.z + a3 * b3.z);
  out[i + 3] = (__bf16)(a0 * b0.w + a1 * b1.w + a2 * b2.w + a3 * b3.w);
}

// ---------------------------------------------------------------------------
// 2) RMSNorm + affine: out = rmsnorm(x,w)*gamma + beta  -> bf16
// ---------------------------------------------------------------------------
__global__ __launch_bounds__(256) void rmsnorm_affine(
    const float* __restrict__ x, const float* __restrict__ w,
    const float* __restrict__ gamma, const float* __restrict__ beta,
    __bf16* __restrict__ out, int D) {
  const int row = blockIdx.x, tid = threadIdx.x;
  const float* xr = x + (size_t)row * D;
  float ss = 0.f;
  for (int i = tid; i < D; i += 256) { float v = xr[i]; ss += v * v; }
  __shared__ float red[256];
  red[tid] = ss;
  __syncthreads();
  for (int s = 128; s > 0; s >>= 1) {
    if (tid < s) red[tid] += red[tid + s];
    __syncthreads();
  }
  const float rs = rsqrtf(red[0] / (float)D + 1.1920929e-07f);
  __bf16* orow = out + (size_t)row * D;
  for (int i = tid; i < D; i += 256)
    orow[i] = (__bf16)(xr[i] * rs * w[i] * gamma[i] + beta[i]);
}

// ---------------------------------------------------------------------------
// 3) WMMA GEMM: C[M,N] = A[M,K] * B[N,K]^T, fused epilogues.
//    64x64 workgroup tile, 8 waves, each wave 16x32, K-step 32.
//    Double-buffered LDS staged via global_load_async_to_lds_b128; previous
//    tile completion enforced with s_wait_asynccnt while next-tile loads
//    stay in flight across the whole compute phase.
//    EPI 0: store bf16               EPI 1: outF = resid + acc
//    EPI 2: dual-B silu(g)*u -> bf16 EPI 3: outF = x0 + (xattn+acc-x0)*scale
// ---------------------------------------------------------------------------
template <int EPI>
__global__ __launch_bounds__(256) void gemm_wmma(
    const __bf16* __restrict__ A, const __bf16* __restrict__ B,
    const __bf16* __restrict__ B2, int Kdim, int N,
    float* __restrict__ outF, __bf16* __restrict__ outH,
    const float* __restrict__ resid, const float* __restrict__ x0,
    const float* __restrict__ xattn, const float* __restrict__ scale_p) {
  constexpr int LD = 48;  // row stride in elems (96B: 16B-aligned, spreads banks)
  __shared__ __bf16 sA[2][64 * LD];
  __shared__ __bf16 sB[2][64 * LD];
  __shared__ __bf16 sB2[(EPI == 2) ? 2 : 1][(EPI == 2) ? 64 * LD : 8];
  const int tid = threadIdx.x;
  const int w = tid >> 5, lane = tid & 31;
  const int wm = w >> 1, wn = w & 1;
  const int lr = lane & 15, kb8 = (lane >> 4) << 3;
  const int m0 = blockIdx.x * 64, n0 = blockIdx.y * 64;
  const int row = tid >> 2, ch = (tid & 3) << 3;

  const __bf16* gA = &A[(size_t)(m0 + row) * Kdim + ch];
  const __bf16* gB = &B[(size_t)(n0 + row) * Kdim + ch];
  const __bf16* gB2 = (EPI == 2) ? &B2[(size_t)(n0 + row) * Kdim + ch] : nullptr;
  const int ldsoff = row * LD + ch;

  // Prologue: stage tile 0 into buffer 0.
  async_ld16(gA, &sA[0][ldsoff]);
  async_ld16(gB, &sB[0][ldsoff]);
  if constexpr (EPI == 2) async_ld16(gB2, &sB2[0][ldsoff]);

  v8f acc[2] = {};
  v8f accu[2] = {};
  const int niter = Kdim >> 5;
  for (int i = 0; i < niter; ++i) {
    const int cur = i & 1, nxt = cur ^ 1;
    if (i + 1 < niter) {  // stage next tile into the other buffer
      const int kn = (i + 1) << 5;
      async_ld16(gA + kn, &sA[nxt][ldsoff]);
      async_ld16(gB + kn, &sB[nxt][ldsoff]);
      if constexpr (EPI == 2) async_ld16(gB2 + kn, &sB2[nxt][ldsoff]);
      if constexpr (EPI == 2) WAIT_ASYNC(0x3); else WAIT_ASYNC(0x2);
    } else {
      WAIT_ASYNC(0x0);
    }
    __syncthreads();
    v16bf af = ld_frag(&sA[cur][(wm * 16 + lr) * LD + kb8]);
#pragma unroll
    for (int j = 0; j < 2; ++j) {
      v16bf bfr = ld_frag(&sB[cur][(wn * 32 + j * 16 + lr) * LD + kb8]);
      acc[j] = WMMA_BF16(af, bfr, acc[j]);
      if constexpr (EPI == 2) {
        v16bf bu = ld_frag(&sB2[cur][(wn * 32 + j * 16 + lr) * LD + kb8]);
        accu[j] = WMMA_BF16(af, bu, accu[j]);
      }
    }
    __syncthreads();  // readers done before this buffer is re-staged
  }
  const int half8 = kb8;  // C frag: row = r + 8*(lane>=16), col = lane&15
  const float sc = (EPI == 3) ? scale_p[0] : 0.f;
#pragma unroll
  for (int j = 0; j < 2; ++j) {
#pragma unroll
    for (int r = 0; r < 8; ++r) {
      const size_t gi =
          (size_t)(m0 + wm * 16 + r + half8) * N + (n0 + wn * 32 + j * 16 + lr);
      const float v = acc[j][r];
      if constexpr (EPI == 0) outH[gi] = (__bf16)v;
      if constexpr (EPI == 1) outF[gi] = resid[gi] + v;
      if constexpr (EPI == 2) {
        const float g = v, u = accu[j][r];
        outH[gi] = (__bf16)((g / (1.f + __expf(-g))) * u);
      }
      if constexpr (EPI == 3) {
        const float xv = x0[gi];
        const float bo = xattn[gi] + v;
        outF[gi] = xv + (bo - xv) * sc;
      }
    }
  }
}

// ---------------------------------------------------------------------------
// 4) Causal flash attention, HD=64. qkv bf16 [T, 3*1024], col = which*1024 +
//    head*64 + hd. Grid: (T/128, H); 8 waves, each owns 16 query rows.
//    K tile staged with async global->LDS; V staged transposed via VGPRs.
// ---------------------------------------------------------------------------
__global__ __launch_bounds__(256) void flash_attn(
    const __bf16* __restrict__ qkv, __bf16* __restrict__ outH) {
  constexpr int LDQ = 3072;
  const int hh = blockIdx.y;
  const int qb = blockIdx.x;
  const int tid = threadIdx.x, w = tid >> 5, lane = tid & 31;
  const int lr = lane & 15, kb8 = (lane >> 4) << 3, half8 = kb8;
  const int m0 = qb * 128 + w * 16;

  __shared__ __bf16 sK[32 * 64];     // [key][hd]
  __shared__ __bf16 sV[64 * 32];     // transposed: [hd][key]
  __shared__ __bf16 sP[8][16 * 32];  // per-wave P scratch [row][key]

  // Q fragments (K-dim = hd, two 32-wide steps) resident in VGPRs
  const __bf16* qbase = qkv + (size_t)(m0 + lr) * LDQ + hh * 64;
  const v16bf qf0 = ld_frag(qbase + kb8);
  const v16bf qf1 = ld_frag(qbase + 32 + kb8);

  v8f o[4] = {};
  float mrun[8], lrun[8];
#pragma unroll
  for (int r = 0; r < 8; ++r) { mrun[r] = -1e30f; lrun[r] = 0.f; }

  const int nkb = qb * 4 + 4;  // causal: keys up to end of this query block
  for (int kb = 0; kb < nkb; ++kb) {
    const int k0 = kb * 32;
    {  // K tile: async DMA to LDS; V tile: transposed scatter via VGPRs
      const int key = tid >> 3, hd = (tid & 7) << 3;
      async_ld16(qkv + (size_t)(k0 + key) * LDQ + 1024 + hh * 64 + hd,
                 &sK[key * 64 + hd]);
      union { v4i q; __bf16 h[8]; } vv;
      vv.q = *(const v4i*)(qkv + (size_t)(k0 + key) * LDQ + 2048 + hh * 64 + hd);
#pragma unroll
      for (int i = 0; i < 8; ++i) sV[(hd + i) * 32 + key] = vv.h[i];
    }
    WAIT_ASYNC(0x0);  // async K writes landed (ds stores waited by barrier)
    __syncthreads();
    if (k0 <= m0 + 15) {  // wave-uniform causal skip
      v8f s0 = {}, s1 = {};
      {
        const __bf16* kp = &sK[lr * 64 + kb8];
        s0 = WMMA_BF16(qf0, ld_frag(kp), s0);
        s0 = WMMA_BF16(qf1, ld_frag(kp + 32), s0);
      }
      {
        const __bf16* kp = &sK[(16 + lr) * 64 + kb8];
        s1 = WMMA_BF16(qf0, ld_frag(kp), s1);
        s1 = WMMA_BF16(qf1, ld_frag(kp + 32), s1);
      }
#pragma unroll
      for (int r = 0; r < 8; ++r) {
        const int grow = m0 + r + half8;
        float a0 = s0[r] * 0.125f;  // 1/sqrt(64)
        float a1 = s1[r] * 0.125f;
        if (k0 + lr > grow) a0 = -1e30f;
        if (k0 + 16 + lr > grow) a1 = -1e30f;
        const float mx = rmax16(fmaxf(a0, a1));
        const float mnew = fmaxf(mrun[r], mx);
        const float fac = __expf(mrun[r] - mnew);
        const float e0 = __expf(a0 - mnew);
        const float e1 = __expf(a1 - mnew);
        const float sum = radd16(e0 + e1);
        lrun[r] = lrun[r] * fac + sum;
        mrun[r] = mnew;
#pragma unroll
        for (int n = 0; n < 4; ++n) o[n][r] *= fac;
        sP[w][(r + half8) * 32 + lr] = (__bf16)e0;
        sP[w][(r + half8) * 32 + 16 + lr] = (__bf16)e1;
      }
      // C-layout -> A-layout via wave-local LDS; DS is in-order per wave,
      // fence keeps the compiler from reordering the cross-lane round trip.
      asm volatile("s_wait_dscnt 0" ::: "memory");
      const v16bf pf = ld_frag(&sP[w][lr * 32 + kb8]);
#pragma unroll
      for (int n = 0; n < 4; ++n) {
        const v16bf vf = ld_frag(&sV[(n * 16 + lr) * 32 + kb8]);
        o[n] = WMMA_BF16(pf, vf, o[n]);
      }
    }
    __syncthreads();
  }
#pragma unroll
  for (int n = 0; n < 4; ++n) {
#pragma unroll
    for (int r = 0; r < 8; ++r) {
      const float val = o[n][r] / lrun[r];
      outH[(size_t)(m0 + r + half8) * 1024 + hh * 64 + n * 16 + lr] = (__bf16)val;
    }
  }
}

// ---------------------------------------------------------------------------
// Launch
// ---------------------------------------------------------------------------
extern "C" void kernel_launch(void* const* d_in, const int* in_sizes, int n_in,
                              void* d_out, int out_size, void* d_ws, size_t ws_size,
                              hipStream_t stream) {
  const float* x          = (const float*)d_in[0];
  const float* gamma      = (const float*)d_in[1];
  const float* beta       = (const float*)d_in[2];
  const float* iter_scale = (const float*)d_in[3];
  const float* qkv_bases  = (const float*)d_in[4];
  const float* o_bases    = (const float*)d_in[5];
  const float* gate_bases = (const float*)d_in[6];
  const float* up_bases   = (const float*)d_in[7];
  const float* down_bases = (const float*)d_in[8];
  const float* a_qkv      = (const float*)d_in[9];
  const float* a_o        = (const float*)d_in[10];
  const float* a_gate     = (const float*)d_in[11];
  const float* a_up       = (const float*)d_in[12];
  const float* a_down     = (const float*)d_in[13];
  const float* norm1_w    = (const float*)d_in[14];
  const float* norm2_w    = (const float*)d_in[15];
  const int*   layer      = (const int*)d_in[16];

  const int Tn = 2048, Dn = 1024, FFn = 4096, Hn = 16;

  char* p = (char*)d_ws;
  auto take = [&](size_t bytes) -> void* {
    void* r = (void*)p;
    p += (bytes + 255) & ~(size_t)255;
    return r;
  };
  __bf16* Wqkv = (__bf16*)take((size_t)3 * Dn * Dn * 2);
  __bf16* Wo   = (__bf16*)take((size_t)Dn * Dn * 2);
  __bf16* Wg   = (__bf16*)take((size_t)FFn * Dn * 2);
  __bf16* Wu   = (__bf16*)take((size_t)FFn * Dn * 2);
  __bf16* Wd   = (__bf16*)take((size_t)Dn * FFn * 2);
  __bf16* h1   = (__bf16*)take((size_t)Tn * Dn * 2);
  __bf16* qkvb = (__bf16*)take((size_t)Tn * 3 * Dn * 2);
  __bf16* attn = (__bf16*)take((size_t)Tn * Dn * 2);
  float*  xat  = (float*)take((size_t)Tn * Dn * 4);
  __bf16* h2   = (__bf16*)take((size_t)Tn * Dn * 2);
  __bf16* act  = (__bf16*)take((size_t)Tn * FFn * 2);

  auto cgrid = [](int n) { return dim3((unsigned)(n / 4 / 256)); };
  combine4<<<cgrid(3 * Dn * Dn), 256, 0, stream>>>(qkv_bases, a_qkv, layer, Wqkv, 3 * Dn * Dn);
  combine4<<<cgrid(Dn * Dn), 256, 0, stream>>>(o_bases, a_o, layer, Wo, Dn * Dn);
  combine4<<<cgrid(FFn * Dn), 256, 0, stream>>>(gate_bases, a_gate, layer, Wg, FFn * Dn);
  combine4<<<cgrid(FFn * Dn), 256, 0, stream>>>(up_bases, a_up, layer, Wu, FFn * Dn);
  combine4<<<cgrid(Dn * FFn), 256, 0, stream>>>(down_bases, a_down, layer, Wd, Dn * FFn);

  rmsnorm_affine<<<Tn, 256, 0, stream>>>(x, norm1_w, gamma, beta, h1, Dn);

  gemm_wmma<0><<<dim3(Tn / 64, 3 * Dn / 64), 256, 0, stream>>>(
      h1, Wqkv, nullptr, Dn, 3 * Dn, nullptr, qkvb, nullptr, nullptr, nullptr, nullptr);

  flash_attn<<<dim3(Tn / 128, Hn), 256, 0, stream>>>(qkvb, attn);

  gemm_wmma<1><<<dim3(Tn / 64, Dn / 64), 256, 0, stream>>>(
      attn, Wo, nullptr, Dn, Dn, xat, nullptr, x, nullptr, nullptr, nullptr);

  rmsnorm_affine<<<Tn, 256, 0, stream>>>(xat, norm2_w, gamma, beta, h2, Dn);

  gemm_wmma<2><<<dim3(Tn / 64, FFn / 64), 256, 0, stream>>>(
      h2, Wg, Wu, Dn, FFn, nullptr, act, nullptr, nullptr, nullptr, nullptr);

  gemm_wmma<3><<<dim3(Tn / 64, Dn / 64), 256, 0, stream>>>(
      act, Wd, nullptr, FFn, Dn, (float*)d_out, nullptr, nullptr, x, xat, iter_scale);
}
